// TransformerEncoderLayer_54941221651328
// MI455X (gfx1250) — matrix operand order repaired and checked by reading, not compiled, not run on hardware
//
#include <hip/hip_runtime.h>
#include <hip/hip_bf16.h>

// ---------------------------------------------------------------------------
// Transformer encoder layer on gfx1250 (MI455X), WMMA f16 path.
//   B=2, T=2048, C=1024, H=16, D=64, FFN=4096.  M = B*T = 4096 rows.
// All GEMMs: f16 A (activations, row-major MxK) x f16 Bt (weights stored
// transposed, NxK row-major) -> f32 accum via v_wmma_f32_16x16x32_f16.
// ---------------------------------------------------------------------------

typedef __attribute__((ext_vector_type(8)))  _Float16 h8;
typedef __attribute__((ext_vector_type(16))) _Float16 v16h;
typedef __attribute__((ext_vector_type(8)))  float    v8f;

#define EMB   1024
#define HEADS 16
#define HDIM  64
#define FFN   4096
#define SEQ   2048
#define BATCH 2
#define MROWS (BATCH * SEQ)   // 4096
#define C3    (3 * EMB)       // 3072

static __device__ __forceinline__ v16h frag16(const _Float16* lo, const _Float16* hi) {
    h8 a = *(const h8*)lo;
    h8 b = *(const h8*)hi;
    return __builtin_shufflevector(a, b, 0,1,2,3,4,5,6,7,8,9,10,11,12,13,14,15);
}

static __device__ __forceinline__ v8f wmma_f16(v16h a, v16h b, v8f c) {
    // D = A(16x32 f16) x B(32x16 f16) + C(16x16 f32)
    return __builtin_amdgcn_wmma_f32_16x16x32_f16(false, a, false, b, (short)0, c, false, false);
}

// ---------------------------------------------------------------------------
// Weight pack: fp32 W[K][N] -> f16 Wt[N][K]   (so B-operand K is contiguous)
// ---------------------------------------------------------------------------
__global__ __launch_bounds__(256) void pack_wt_k(const float* __restrict__ src,
                                                 _Float16* __restrict__ dst,
                                                 int K, int N) {
    size_t idx = (size_t)blockIdx.x * 256 + threadIdx.x;
    size_t tot = (size_t)K * N;
    if (idx >= tot) return;
    size_t n = idx / K, k = idx % K;
    dst[idx] = (_Float16)src[k * (size_t)N + n];
}

__global__ __launch_bounds__(256) void copy_f32_k(const float* __restrict__ src,
                                                  float* __restrict__ dst, int n) {
    int i = blockIdx.x * 256 + threadIdx.x;
    if (i < n) dst[i] = src[i];
}

// ---------------------------------------------------------------------------
// LayerNorm over rows of EMB; optional f32 copy (residual base) + f16 copy.
// ---------------------------------------------------------------------------
__global__ __launch_bounds__(256) void layernorm_k(const float* __restrict__ x,
                                                   const float* __restrict__ g,
                                                   const float* __restrict__ b,
                                                   float* __restrict__ outF,
                                                   _Float16* __restrict__ outH) {
    const int row = blockIdx.x;
    const float* xr = x + (size_t)row * EMB;
    __shared__ float s1[256], s2[256];
    float sum = 0.f, sq = 0.f;
    for (int c = threadIdx.x; c < EMB; c += 256) {
        float v = xr[c];
        sum += v; sq += v * v;
    }
    s1[threadIdx.x] = sum; s2[threadIdx.x] = sq;
    __syncthreads();
    for (int off = 128; off > 0; off >>= 1) {
        if (threadIdx.x < off) {
            s1[threadIdx.x] += s1[threadIdx.x + off];
            s2[threadIdx.x] += s2[threadIdx.x + off];
        }
        __syncthreads();
    }
    const float mu  = s1[0] * (1.0f / EMB);
    const float var = s2[0] * (1.0f / EMB) - mu * mu;
    const float rs  = rsqrtf(var + 1e-5f);
    for (int c = threadIdx.x; c < EMB; c += 256) {
        float y = (xr[c] - mu) * rs * g[c] + b[c];
        if (outF) outF[(size_t)row * EMB + c] = y;
        outH[(size_t)row * EMB + c] = (_Float16)y;
    }
}

// ---------------------------------------------------------------------------
// WMMA GEMM: C[M,N] = A[M,K] @ Bt[N,K]^T + bias, epilogue by MODE.
//   MODE 0: store f16
//   MODE 1: exact-erf GELU then store f16
//   MODE 2: add residual (f32), store f32
// Block = 256 threads = 8 waves; wave tile 16x64; block tile 128x64.
// ---------------------------------------------------------------------------
template <int MODE>
__global__ __launch_bounds__(256) void gemm_wmma_k(const _Float16* __restrict__ A,
                                                   const _Float16* __restrict__ Bt,
                                                   const float* __restrict__ bias,
                                                   const float* __restrict__ res,
                                                   float* __restrict__ outF,
                                                   _Float16* __restrict__ outH,
                                                   int M, int N, int K) {
    const int lane = threadIdx.x & 31;
    const int wave = threadIdx.x >> 5;
    const int hi   = lane >> 4;       // half-wave group
    const int ln   = lane & 15;
    const int m0   = blockIdx.y * 128 + wave * 16;
    const int n0   = blockIdx.x * 64;

    const _Float16* aRow = A + (size_t)(m0 + ln) * K;
    const _Float16* bRow0 = Bt + (size_t)(n0 + ln) * K;
    const _Float16* bRow1 = Bt + (size_t)(n0 + 16 + ln) * K;
    const _Float16* bRow2 = Bt + (size_t)(n0 + 32 + ln) * K;
    const _Float16* bRow3 = Bt + (size_t)(n0 + 48 + ln) * K;

    v8f acc[4];
    const v8f zero = {};
    acc[0] = zero; acc[1] = zero; acc[2] = zero; acc[3] = zero;

    for (int k = 0; k < K; k += 32) {
        // A fragment: lanes 0-15 row=ln K={k..k+15};
        //             lanes 16-31 same row-set, K shifted by 16.
        v16h av = frag16(aRow + k + 16 * hi, aRow + k + 16 * hi + 8);
        __builtin_prefetch(aRow + k + 64, 0, 1);          // global_prefetch_b8
        // B fragments: lane = output column, 16 contiguous K each half-group.
        const int ko = k + 16 * hi;
        v16h b0 = frag16(bRow0 + ko, bRow0 + ko + 8);
        v16h b1 = frag16(bRow1 + ko, bRow1 + ko + 8);
        v16h b2 = frag16(bRow2 + ko, bRow2 + ko + 8);
        v16h b3 = frag16(bRow3 + ko, bRow3 + ko + 8);
        __builtin_prefetch(bRow0 + ko + 64, 0, 1);
        acc[0] = wmma_f16(av, b0, acc[0]);
        acc[1] = wmma_f16(av, b1, acc[1]);
        acc[2] = wmma_f16(av, b2, acc[2]);
        acc[3] = wmma_f16(av, b3, acc[3]);
    }

#pragma unroll
    for (int j = 0; j < 4; ++j) {
        const int c = n0 + j * 16 + ln;
        const float bv = bias[c];
#pragma unroll
        for (int v = 0; v < 8; ++v) {
            const int r = m0 + v + 8 * hi;
            float val = acc[j][v] + bv;
            if (MODE == 0) {
                outH[(size_t)r * N + c] = (_Float16)val;
            } else if (MODE == 1) {
                val = 0.5f * val * (1.0f + erff(val * 0.70710678118654752f));
                outH[(size_t)r * N + c] = (_Float16)val;
            } else {
                outF[(size_t)r * N + c] = val + res[(size_t)r * N + c];
            }
        }
    }
}

// ---------------------------------------------------------------------------
// V transpose: qkv16[(b*T+t)*3072 + 2048 + h*64 + d] -> vt[((b*H+h)*64+d)*T + t]
// ---------------------------------------------------------------------------
__global__ __launch_bounds__(256) void vtrans_k(const _Float16* __restrict__ qkv,
                                                _Float16* __restrict__ vt) {
    size_t idx = (size_t)blockIdx.x * 256 + threadIdx.x;
    if (idx >= (size_t)BATCH * HEADS * HDIM * SEQ) return;
    int t = idx & (SEQ - 1);
    size_t r = idx >> 11;
    int d = r & 63; r >>= 6;
    int h = r & 15;
    int b = (int)(r >> 4);
    vt[idx] = qkv[((size_t)(b * SEQ + t)) * C3 + 2 * EMB + h * HDIM + d];
}

// ---------------------------------------------------------------------------
// Flash-style attention: one wave per (b,h, 16-query block).
//   S = Q Kt (4 WMMA per 32-key block), online softmax, O += P V (4 WMMA).
// ---------------------------------------------------------------------------
__global__ __launch_bounds__(32) void attention_k(const _Float16* __restrict__ qkv,
                                                  const _Float16* __restrict__ vt,
                                                  _Float16* __restrict__ attn) {
    const int bh = blockIdx.x;                 // 0..31
    const int b  = bh >> 4;
    const int h  = bh & 15;
    const int m0 = blockIdx.y * 16;            // query block within sequence
    const int lane = threadIdx.x;
    const int hi = lane >> 4, ln = lane & 15;

    __shared__ _Float16 P[16][40];             // padded 16x32 P tile

    // Q fragments for d=0..31 and d=32..63 (held for the whole key loop).
    const _Float16* qb = qkv + ((size_t)(b * SEQ + m0 + ln)) * C3 + h * HDIM;
    const v16h q0 = frag16(qb +      16 * hi, qb +      16 * hi + 8);
    const v16h q1 = frag16(qb + 32 + 16 * hi, qb + 32 + 16 * hi + 8);

    float Mrow[8], Lrow[8];
    v8f O[4];
    const v8f zero = {};
#pragma unroll
    for (int v = 0; v < 8; ++v) { Mrow[v] = -3.0e38f; Lrow[v] = 0.f; }
#pragma unroll
    for (int j = 0; j < 4; ++j) O[j] = zero;

    const float scale = 0.125f;                // 1/sqrt(64)

    for (int s0 = 0; s0 < SEQ; s0 += 32) {
        // ---- S = Q @ K^T for 32 keys (two 16-key N-tiles) ----
        v8f s0t, s1t;
        {
            const _Float16* kb0 = qkv + ((size_t)(b * SEQ + s0 + ln)) * C3 + EMB + h * HDIM;
            const _Float16* kb1 = qkv + ((size_t)(b * SEQ + s0 + 16 + ln)) * C3 + EMB + h * HDIM;
            v16h kA0 = frag16(kb0 + 16 * hi,      kb0 + 16 * hi + 8);
            v16h kA1 = frag16(kb0 + 32 + 16 * hi, kb0 + 32 + 16 * hi + 8);
            v16h kB0 = frag16(kb1 + 16 * hi,      kb1 + 16 * hi + 8);
            v16h kB1 = frag16(kb1 + 32 + 16 * hi, kb1 + 32 + 16 * hi + 8);
            v8f t0 = zero, t1 = zero;
            t0 = wmma_f16(q0, kA0, t0); t0 = wmma_f16(q1, kA1, t0);
            t1 = wmma_f16(q0, kB0, t1); t1 = wmma_f16(q1, kB1, t1);
            s0t = t0; s1t = t1;
        }

        // ---- online softmax over the 32 keys (row-wise across 16 lanes) ----
#pragma unroll
        for (int v = 0; v < 8; ++v) {
            float a0 = s0t[v] * scale;
            float a1 = s1t[v] * scale;
            float mx = fmaxf(a0, a1);
            mx = fmaxf(mx, __shfl_xor(mx, 1));
            mx = fmaxf(mx, __shfl_xor(mx, 2));
            mx = fmaxf(mx, __shfl_xor(mx, 4));
            mx = fmaxf(mx, __shfl_xor(mx, 8));
            const float nm = fmaxf(Mrow[v], mx);
            const float alpha = __expf(Mrow[v] - nm);
            const float p0 = __expf(a0 - nm);
            const float p1 = __expf(a1 - nm);
            float rs = p0 + p1;
            rs += __shfl_xor(rs, 1);
            rs += __shfl_xor(rs, 2);
            rs += __shfl_xor(rs, 4);
            rs += __shfl_xor(rs, 8);
            Lrow[v] = Lrow[v] * alpha + rs;
            Mrow[v] = nm;
#pragma unroll
            for (int j = 0; j < 4; ++j) O[j][v] *= alpha;
            // C-layout -> LDS (row = v + 8*hi, col = key)
            P[v + 8 * hi][ln]      = (_Float16)p0;
            P[v + 8 * hi][16 + ln] = (_Float16)p1;
        }
        __syncthreads();
        // re-read as A-operand layout (lane = row, 16 contiguous K per half-wave)
        v16h pf = frag16(&P[ln][16 * hi], &P[ln][16 * hi + 8]);
        __syncthreads();

        // ---- O += P(16x32) @ V(32x64), four 16-wide d tiles ----
#pragma unroll
        for (int j = 0; j < 4; ++j) {
            const _Float16* vb = vt + ((size_t)(bh * HDIM + j * 16 + ln)) * SEQ + s0 + 16 * hi;
            v16h vf = frag16(vb, vb + 8);
            O[j] = wmma_f16(pf, vf, O[j]);
        }
    }

    // ---- finalize: O / L, store f16 into [M, EMB] ----
#pragma unroll
    for (int v = 0; v < 8; ++v) {
        const float inv = 1.0f / Lrow[v];
        const size_t r = (size_t)(b * SEQ + m0 + v + 8 * hi);
#pragma unroll
        for (int j = 0; j < 4; ++j)
            attn[r * EMB + h * HDIM + j * 16 + ln] = (_Float16)(O[j][v] * inv);
    }
}

// ---------------------------------------------------------------------------
// Host-side orchestration
// ---------------------------------------------------------------------------
extern "C" void kernel_launch(void* const* d_in, const int* in_sizes, int n_in,
                              void* d_out, int out_size, void* d_ws, size_t ws_size,
                              hipStream_t stream) {
    (void)in_sizes; (void)n_in; (void)out_size; (void)ws_size;
    const float* x     = (const float*)d_in[0];
    const float* ln1_g = (const float*)d_in[2];
    const float* ln1_b = (const float*)d_in[3];
    const float* ln2_g = (const float*)d_in[4];
    const float* ln2_b = (const float*)d_in[5];
    const float* Wq = (const float*)d_in[6];
    const float* bq = (const float*)d_in[7];
    const float* Wk = (const float*)d_in[8];
    const float* bk = (const float*)d_in[9];
    const float* Wv = (const float*)d_in[10];
    const float* bv = (const float*)d_in[11];
    const float* Wo = (const float*)d_in[12];
    const float* bo = (const float*)d_in[13];
    const float* W1 = (const float*)d_in[14];
    const float* b1 = (const float*)d_in[15];
    const float* W2 = (const float*)d_in[16];
    const float* b2 = (const float*)d_in[17];
    float* out = (float*)d_out;

    char* ws = (char*)d_ws;
    float*    xnF   = (float*)   (ws + 0);                         // 16 MB
    _Float16* xn16  = (_Float16*)(ws + 16777216);                  //  8 MB
    _Float16* wqkvt = (_Float16*)(ws + 25165824);                  //  6 MB
    float*    bqkv  = (float*)   (ws + 31457280);                  // 12 KB
    _Float16* wot   = (_Float16*)(ws + 31469568);                  //  2 MB
    _Float16* w1t   = (_Float16*)(ws + 33566720);                  //  8 MB
    _Float16* w2t   = (_Float16*)(ws + 41955328);                  //  8 MB
    _Float16* qkv16 = (_Float16*)(ws + 50343936);                  // 24 MB
    _Float16* vt    = (_Float16*)(ws + 75509760);                  //  8 MB
    _Float16* at16  = (_Float16*)(ws + 83898368);                  //  8 MB
    float*    x2    = (float*)   (ws + 92286976);                  // 16 MB
    _Float16* h16   = (_Float16*)(ws + 109064192);                 //  8 MB
    _Float16* h1_16 = (_Float16*)(ws + 117452800);                 // 32 MB

    const int TB = 256;
    const size_t e1m = (size_t)EMB * EMB;          // 1M elems
    const int g1m = (int)((e1m + TB - 1) / TB);
    const size_t e4m = (size_t)EMB * FFN;          // 4M elems
    const int g4m = (int)((e4m + TB - 1) / TB);

    // 1) weight packs (fp32 -> f16 transposed)
    pack_wt_k<<<g1m, TB, 0, stream>>>(Wq, wqkvt,            EMB, EMB);
    pack_wt_k<<<g1m, TB, 0, stream>>>(Wk, wqkvt + e1m,      EMB, EMB);
    pack_wt_k<<<g1m, TB, 0, stream>>>(Wv, wqkvt + 2 * e1m,  EMB, EMB);
    pack_wt_k<<<g1m, TB, 0, stream>>>(Wo, wot,              EMB, EMB);
    pack_wt_k<<<g4m, TB, 0, stream>>>(W1, w1t,              EMB, FFN);
    pack_wt_k<<<g4m, TB, 0, stream>>>(W2, w2t,              FFN, EMB);
    copy_f32_k<<<4, TB, 0, stream>>>(bq, bqkv,            EMB);
    copy_f32_k<<<4, TB, 0, stream>>>(bk, bqkv + EMB,      EMB);
    copy_f32_k<<<4, TB, 0, stream>>>(bv, bqkv + 2 * EMB,  EMB);

    // 2) ln1: x -> xnF (residual base, per reference semantics) + xn16
    layernorm_k<<<MROWS, TB, 0, stream>>>(x, ln1_g, ln1_b, xnF, xn16);

    // 3) fused QKV GEMM: [4096,1024] x [1024,3072] -> qkv16 f16
    gemm_wmma_k<0><<<dim3(C3 / 64, MROWS / 128), TB, 0, stream>>>(
        xn16, wqkvt, bqkv, nullptr, nullptr, qkv16, MROWS, C3, EMB);

    // 4) V transpose for the PV WMMA B-operand
    vtrans_k<<<(int)(((size_t)BATCH * HEADS * HDIM * SEQ + TB - 1) / TB), TB, 0, stream>>>(qkv16, vt);

    // 5) attention (flash-style, WMMA for S and PV)
    attention_k<<<dim3(BATCH * HEADS, SEQ / 16), 32, 0, stream>>>(qkv16, vt, at16);

    // 6) output proj + residual: x2 = xnF + attn @ Wo + bo
    gemm_wmma_k<2><<<dim3(EMB / 64, MROWS / 128), TB, 0, stream>>>(
        at16, wot, bo, xnF, x2, nullptr, MROWS, EMB, EMB);

    // 7) ln2: x2 -> h16
    layernorm_k<<<MROWS, TB, 0, stream>>>(x2, ln2_g, ln2_b, nullptr, h16);

    // 8) FFN1 + exact GELU: h1_16 = gelu(h16 @ W1 + b1)
    gemm_wmma_k<1><<<dim3(FFN / 64, MROWS / 128), TB, 0, stream>>>(
        h16, w1t, b1, nullptr, nullptr, h1_16, MROWS, FFN, EMB);

    // 9) FFN2 + residual: out = x2 + h1 @ W2 + b2
    gemm_wmma_k<2><<<dim3(EMB / 64, MROWS / 128), TB, 0, stream>>>(
        h1_16, w2t, b2, x2, out, nullptr, MROWS, EMB, FFN);
}
